// MultiHeadAttention_KVCache_6485400617217
// MI455X (gfx1250) — compile-verified, matrix operand order checked by reading, not run on hardware
//
#include <hip/hip_runtime.h>

// ---------------------------------------------------------------------------
// MI455X (gfx1250) multi-head attention w/ KV cache, bf16 WMMA pipeline.
// Stages: fp32->bf16 prep, QKV GEMM (WMMA + TDM staging),
//         flash attention (WMMA + TDM staging), proj GEMM.
// ---------------------------------------------------------------------------

typedef __bf16 v16bf __attribute__((ext_vector_type(16)));
typedef __bf16 v8bf  __attribute__((ext_vector_type(8)));
typedef float  v8f   __attribute__((ext_vector_type(8)));
typedef unsigned v4u __attribute__((ext_vector_type(4)));
typedef int      v8i __attribute__((ext_vector_type(8)));
typedef int      v4i __attribute__((ext_vector_type(4)));

constexpr int B_      = 2;
constexpr int TQ_     = 1024;
constexpr int PREFIX_ = 1024;
constexpr int H_      = 16;
constexpr int HD_     = 128;
constexpr int D_      = H_ * HD_;        // 2048
constexpr int TK_     = PREFIX_ + TQ_;   // 2048
constexpr int MROWS_  = B_ * TQ_;        // 2048

__device__ __forceinline__ __bf16 f2bf(float f) {
  unsigned u = __builtin_bit_cast(unsigned, f);
  u += 0x7FFFu + ((u >> 16) & 1u);               // round-to-nearest-even
  unsigned short s = (unsigned short)(u >> 16);
  return __builtin_bit_cast(__bf16, s);
}

__device__ __forceinline__ v8f wmma_bf16(v16bf a, v16bf b, v8f c) {
  // D(16x16,f32) = A(16x32,bf16) x B(32x16,bf16) + C
  return __builtin_amdgcn_wmma_f32_16x16x32_bf16(
      /*neg_a=*/false, a, /*neg_b=*/false, b,
      /*c_mod=*/(short)0, c, /*reuse_a=*/false, /*reuse_b=*/false);
}

// Tensor Data Mover: 2-D tile of 2-byte elements, global -> LDS.
// D# built per CDNA5 ISA (group0 128b + group1 256b,
// groups 2/3 + extra group zero => tile up to 2-D). Tracked on TENSORcnt.
// This toolchain exposes the 6-arg builtin:
//   (uint32x4 g0, int32x8 g1, int32x4, int32x4, int32x8, i32 cpol)
__device__ __forceinline__ void tdm_load_2d_bf16(const void* lds_dst,
                                                 const void* gsrc,
                                                 unsigned tile_w,   // elems
                                                 unsigned tile_h,   // rows
                                                 unsigned stride) { // elems
  // LDS aperture maps flat addr[31:0] directly to LDS byte offset.
  unsigned lds_off = (unsigned)(unsigned long long)lds_dst;
  unsigned long long ga = (unsigned long long)gsrc;
  v4u g0;
  g0[0] = 1u;                                        // count=1, user mode
  g0[1] = lds_off;                                   // lds_addr
  g0[2] = (unsigned)ga;                              // global_addr[31:0]
  g0[3] = (unsigned)((ga >> 32) & 0x1FFFFFFu) | (2u << 30);  // [56:32] | type=2
  // group1 bitfields (bit offsets per ISA):
  //  [17:16] data_size=1 (2B)   [79:48] tensor_dim0   [111:80] tensor_dim1
  //  [127:112] tile_dim0        [143:128] tile_dim1   [207:160] dim0_stride
  v8i g1;
  g1[0] = (int)(1u << 16);                           // data_size = 2 bytes
  g1[1] = (int)((tile_w & 0xFFFFu) << 16);           // tensor_dim0 lo16
  g1[2] = (int)((tile_w >> 16) | ((tile_h & 0xFFFFu) << 16));  // d0 hi / d1 lo
  g1[3] = (int)((tile_h >> 16) | ((tile_w & 0xFFFFu) << 16));  // d1 hi / tile_dim0
  g1[4] = (int)(tile_h & 0xFFFFu);                   // tile_dim1 (tile_dim2=0)
  g1[5] = (int)stride;                               // tensor_dim0_stride[31:0]
  g1[6] = 0;
  g1[7] = 0;
  v4i z4 = {0, 0, 0, 0};
  v8i z8 = {0, 0, 0, 0, 0, 0, 0, 0};
  __builtin_amdgcn_tensor_load_to_lds(g0, g1, z4, z4, z8, 0);
}

// ----------------------------- prep kernels --------------------------------

__global__ void k_cvt_copy(const float* __restrict__ s, __bf16* __restrict__ d, int n) {
  for (int i = blockIdx.x * blockDim.x + threadIdx.x; i < n;
       i += (int)(gridDim.x * blockDim.x))
    d[i] = f2bf(s[i]);
}

// src[k*N + n] -> dst[n*K + k]  (so GEMM B-operand is K-contiguous per column)
__global__ void k_cvt_transpose(const float* __restrict__ s, __bf16* __restrict__ d,
                                int K, int N) {
  int total = K * N;
  for (int i = blockIdx.x * blockDim.x + threadIdx.x; i < total;
       i += (int)(gridDim.x * blockDim.x)) {
    int k = i / N, n = i - k * N;
    d[(size_t)n * K + k] = f2bf(s[i]);
  }
}

// cache [B*H][PREFIX][HD] fp32 -> kv buffer [B*H][TK][HD] bf16, rows [0,PREFIX)
__global__ void k_cvt_cache(const float* __restrict__ s, __bf16* __restrict__ d) {
  int total = B_ * H_ * PREFIX_ * HD_;
  for (int i = blockIdx.x * blockDim.x + threadIdx.x; i < total;
       i += (int)(gridDim.x * blockDim.x)) {
    int dd  = i & (HD_ - 1);
    int rem = i >> 7;                 // / HD_
    int p   = rem & (PREFIX_ - 1);
    int bh  = rem >> 10;              // / PREFIX_
    d[((size_t)bh * TK_ + p) * HD_ + dd] = f2bf(s[i]);
  }
}

// ------------------------------ bf16 GEMM ----------------------------------
// C[M,N] = A[M,K] @ B[K,N], with B supplied transposed (Bt[N][K]).
// Block tile 128x64, BK=32, 256 threads = 8 waves laid out 4(M) x 2(N),
// each wave computes a 32x32 sub-tile via 2x2 WMMA 16x16x32 accumulators.
// Tiles are staged into LDS by the Tensor Data Mover (wave 0 issues,
// completion on TENSORcnt), with per-lane prefetch of the next tiles.
// MODE 0: scatter bf16 result into Q / K-cache-tail / V-cache-tail.
// MODE 1: fp32 out + bias (final projection).

template <int MODE>
__global__ __launch_bounds__(256) void k_gemm(
    const __bf16* __restrict__ A, const __bf16* __restrict__ Bt,
    int M, int N, int K,
    __bf16* __restrict__ Qh, __bf16* __restrict__ Kh, __bf16* __restrict__ Vh,
    const float* __restrict__ bias, float* __restrict__ Out) {
  constexpr int BM = 128, BN = 64, BK = 32;
  __shared__ alignas(32) __bf16 As[BM][BK];
  __shared__ alignas(32) __bf16 Bs[BN][BK];

  const int tid  = threadIdx.x;
  const int lane = tid & 31;
  const int wave = tid >> 5;
  const int wm = wave >> 1, wn = wave & 1;       // 4 x 2 wave grid
  const int lrow = lane & 15, lhi = lane >> 4;   // fragment lane decomposition
  const int m0 = blockIdx.y * BM, n0 = blockIdx.x * BN;

  v8f acc[2][2] = {};

  const int ar = tid >> 1, ac = (tid & 1) * 16;  // per-lane prefetch targets
  const int br = tid >> 2, bc = (tid & 3) * 8;

  for (int k0 = 0; k0 < K; k0 += BK) {
    __syncthreads();                 // previous tile fully consumed
    if (wave == 0) {
      tdm_load_2d_bf16(&As[0][0], &A[(size_t)m0 * K + k0], BK, BM, (unsigned)K);
      tdm_load_2d_bf16(&Bs[0][0], &Bt[(size_t)n0 * K + k0], BK, BN, (unsigned)K);
    }
    // Warm L2 for the next k-step while the TDM pulls this one.
    if (k0 + BK < K) {
      __builtin_prefetch(&A[(size_t)(m0 + ar) * K + k0 + BK + ac], 0, 1);
      __builtin_prefetch(&Bt[(size_t)(n0 + br) * K + k0 + BK + bc], 0, 1);
    }
    if (wave == 0) __builtin_amdgcn_s_wait_tensorcnt(0);
    __syncthreads();                 // tile visible to all waves

    v16bf af[2], bfr[2];
#pragma unroll
    for (int mt = 0; mt < 2; ++mt)
      af[mt] = *(const v16bf*)&As[wm * 32 + mt * 16 + lrow][lhi * 16];
#pragma unroll
    for (int nt = 0; nt < 2; ++nt)
      bfr[nt] = *(const v16bf*)&Bs[wn * 32 + nt * 16 + lrow][lhi * 16];
#pragma unroll
    for (int mt = 0; mt < 2; ++mt)
#pragma unroll
      for (int nt = 0; nt < 2; ++nt)
        acc[mt][nt] = wmma_bf16(af[mt], bfr[nt], acc[mt][nt]);
  }

  // Epilogue: C fragment layout -> (M = r + 8*lhi, N = lrow) per VGPR r.
#pragma unroll
  for (int mt = 0; mt < 2; ++mt)
#pragma unroll
    for (int nt = 0; nt < 2; ++nt)
#pragma unroll
      for (int r = 0; r < 8; ++r) {
        int gm = m0 + wm * 32 + mt * 16 + r + 8 * lhi;
        int gn = n0 + wn * 32 + nt * 16 + lrow;
        float v = acc[mt][nt][r];
        if (MODE == 0) {
          int b = gm >> 10, t = gm & (TQ_ - 1);
          int which = gn >> 11, inner = gn & (D_ - 1);
          int h = inner >> 7, dd = inner & (HD_ - 1);
          __bf16 bv = f2bf(v);
          size_t bhh = (size_t)(b * H_ + h);
          if (which == 0)
            Qh[(bhh * TQ_ + t) * HD_ + dd] = bv;
          else if (which == 1)
            Kh[(bhh * TK_ + PREFIX_ + t) * HD_ + dd] = bv;
          else
            Vh[(bhh * TK_ + PREFIX_ + t) * HD_ + dd] = bv;
        } else {
          Out[(size_t)gm * N + gn] = v + bias[gn];
        }
      }
}

// --------------------------- flash attention -------------------------------
// Block = (b*h, 64-query tile), 128 threads = 4 waves; wave owns 16 q-rows.
// Online softmax over 32-key chunks; prefix fully visible, causal over chunk.
// K chunk staged by TDM (contiguous 8KB tile), V chunk transposed in-flight.

__global__ __launch_bounds__(128) void k_attn(
    const __bf16* __restrict__ Qh, const __bf16* __restrict__ Kh,
    const __bf16* __restrict__ Vh, __bf16* __restrict__ Ctx) {
  constexpr int BQ = 64, SK = 32;
  __shared__ alignas(32) __bf16 Ks[SK][HD_];    // 8 KB
  __shared__ alignas(32) __bf16 Vts[HD_][SK];   // 8 KB (transposed V chunk)
  __shared__ alignas(32) __bf16 Ps[4][16][SK];  // 4 KB (per-wave P tiles)

  const int bh = blockIdx.x, qt = blockIdx.y;
  const int b = bh >> 4, h = bh & 15;
  const int tid = threadIdx.x, wave = tid >> 5, lane = tid & 31;
  const int lrow = lane & 15, lhi = lane >> 4;

  const __bf16* Qg = Qh + ((size_t)bh * TQ_ + qt * BQ) * HD_;
  const __bf16* Kg = Kh + (size_t)bh * TK_ * HD_;
  const __bf16* Vg = Vh + (size_t)bh * TK_ * HD_;

  // Q A-fragments: 16 rows x HD=128 -> 4 fragments of 16x32.
  v16bf aq[4];
  const int qloc = wave * 16 + lrow;
#pragma unroll
  for (int c = 0; c < 4; ++c)
    aq[c] = *(const v16bf*)&Qg[(size_t)qloc * HD_ + c * 32 + lhi * 16];

  v8f O[8] = {};           // 16 x 128 fp32 accumulator (8 n-tiles)
  float mr[8], ls[8];      // per-row running max / sum (row = (r, lane-half))
#pragma unroll
  for (int r = 0; r < 8; ++r) { mr[r] = -1e30f; ls[r] = 0.0f; }
  const float scale = 0.08838834764831845f;  // 1/sqrt(128)

  for (int kt = 0; kt < TK_; kt += SK) {
    __syncthreads();
    // K chunk: one contiguous 4096-elem (8KB) TDM tile into Ks.
    if (wave == 0)
      tdm_load_2d_bf16(&Ks[0][0], &Kg[(size_t)kt * HD_], SK * HD_, 1, SK * HD_);
    // Stage V chunk transposed: thread tid owns column d = tid; loads are
    // coalesced across threads per key row; packed b128 stores to LDS.
    {
      alignas(16) unsigned tmp[16];
#pragma unroll
      for (int k = 0; k < SK; k += 2) {
        unsigned lo = *(const unsigned short*)&Vg[(size_t)(kt + k) * HD_ + tid];
        unsigned hi = *(const unsigned short*)&Vg[(size_t)(kt + k + 1) * HD_ + tid];
        tmp[k >> 1] = lo | (hi << 16);
      }
#pragma unroll
      for (int i = 0; i < 4; ++i)
        *(uint4*)&Vts[tid][i * 8] = *(const uint4*)&tmp[i * 4];
    }
    // Warm L2 for the next chunk (one 64B line per lane covers 8KB each).
    if (kt + SK < TK_) {
      __builtin_prefetch((const char*)&Kg[(size_t)(kt + SK) * HD_] + tid * 64, 0, 1);
      __builtin_prefetch((const char*)&Vg[(size_t)(kt + SK) * HD_] + tid * 64, 0, 1);
    }
    if (wave == 0) __builtin_amdgcn_s_wait_tensorcnt(0);
    __syncthreads();

    // S = Q @ K^T for two 16-key tiles (K-fragments contiguous along HD).
    v8f s0 = {}, s1 = {};
#pragma unroll
    for (int c = 0; c < 4; ++c) {
      v16bf bk0 = *(const v16bf*)&Ks[lrow][c * 32 + lhi * 16];
      v16bf bk1 = *(const v16bf*)&Ks[16 + lrow][c * 32 + lhi * 16];
      s0 = wmma_bf16(aq[c], bk0, s0);
      s1 = wmma_bf16(aq[c], bk1, s1);
    }

    // Mask + online softmax. Row (r, lane-half) stats are lane-local after a
    // 16-lane xor-reduction (halves of the wave32 reduce independently).
    const int qpb = qt * BQ + wave * 16 + 8 * lhi;
#pragma unroll
    for (int r = 0; r < 8; ++r) {
      int qpos = qpb + r;
      int kp0 = kt + lrow, kp1 = kt + 16 + lrow;
      float v0 = (kp0 <= PREFIX_ + qpos) ? s0[r] * scale : -1e30f;
      float v1 = (kp1 <= PREFIX_ + qpos) ? s1[r] * scale : -1e30f;
      float vmax = fmaxf(v0, v1);
#pragma unroll
      for (int off = 1; off < 16; off <<= 1)
        vmax = fmaxf(vmax, __shfl_xor(vmax, off));
      float mn   = fmaxf(mr[r], vmax);
      float corr = __expf(mr[r] - mn);
      mr[r] = mn;
      float p0 = __expf(v0 - mn), p1 = __expf(v1 - mn);
      float rs = p0 + p1;
#pragma unroll
      for (int off = 1; off < 16; off <<= 1) rs += __shfl_xor(rs, off);
      ls[r] = ls[r] * corr + rs;
#pragma unroll
      for (int nt = 0; nt < 8; ++nt) O[nt][r] *= corr;
      // C-layout -> A-layout transpose via per-wave LDS tile.
      Ps[wave][r + 8 * lhi][lrow]      = f2bf(p0);
      Ps[wave][r + 8 * lhi][16 + lrow] = f2bf(p1);
    }
    // Same-wave LDS write->read ordering for the P tile.
    asm volatile("s_wait_dscnt 0" ::: "memory");

    // O += P(16x32) @ V(32x128); V B-fragments contiguous in transposed tile.
    v16bf ap = *(const v16bf*)&Ps[wave][lrow][lhi * 16];
#pragma unroll
    for (int nt = 0; nt < 8; ++nt) {
      v16bf bv = *(const v16bf*)&Vts[nt * 16 + lrow][lhi * 16];
      O[nt] = wmma_bf16(ap, bv, O[nt]);
    }
  }

#pragma unroll
  for (int r = 0; r < 8; ++r) ls[r] = 1.0f / ls[r];
  // Write ctx as [B][TQ][H*HD] bf16 (matches reference head re-interleave).
#pragma unroll
  for (int nt = 0; nt < 8; ++nt)
#pragma unroll
    for (int r = 0; r < 8; ++r) {
      int qpos = qt * BQ + wave * 16 + r + 8 * lhi;
      int dd = nt * 16 + lrow;
      Ctx[((size_t)(b * TQ_ + qpos)) * D_ + h * HD_ + dd] = f2bf(O[nt][r] * ls[r]);
    }
}

// ------------------------------- launcher ----------------------------------

extern "C" void kernel_launch(void* const* d_in, const int* in_sizes, int n_in,
                              void* d_out, int out_size, void* d_ws, size_t ws_size,
                              hipStream_t stream) {
  (void)in_sizes; (void)n_in; (void)out_size; (void)ws_size;
  const float* x      = (const float*)d_in[0];
  const float* cachek = (const float*)d_in[1];
  const float* cachev = (const float*)d_in[2];
  const float* Wqkv   = (const float*)d_in[3];
  const float* Wproj  = (const float*)d_in[4];
  const float* bproj  = (const float*)d_in[5];
  float* out = (float*)d_out;

  char* ws = (char*)d_ws;
  size_t off = 0;
  auto take = [&](size_t elems) -> __bf16* {
    void* p = ws + off;
    off = (off + elems * 2 + 255) & ~(size_t)255;
    return (__bf16*)p;
  };
  __bf16* Xh     = take((size_t)MROWS_ * D_);       // x in bf16
  __bf16* Wqkvt  = take((size_t)D_ * 3 * D_);       // W_qkv^T bf16
  __bf16* Wprojt = take((size_t)D_ * D_);           // W_proj^T bf16
  __bf16* Qh     = take((size_t)B_ * H_ * TQ_ * HD_);
  __bf16* Kh     = take((size_t)B_ * H_ * TK_ * HD_);
  __bf16* Vh     = take((size_t)B_ * H_ * TK_ * HD_);
  __bf16* Ctx    = take((size_t)MROWS_ * D_);

  // 1) fp32 -> bf16 prep
  k_cvt_copy<<<2048, 256, 0, stream>>>(x, Xh, MROWS_ * D_);
  k_cvt_transpose<<<4096, 256, 0, stream>>>(Wqkv, Wqkvt, D_, 3 * D_);
  k_cvt_transpose<<<2048, 256, 0, stream>>>(Wproj, Wprojt, D_, D_);
  k_cvt_cache<<<2048, 256, 0, stream>>>(cachek, Kh);
  k_cvt_cache<<<2048, 256, 0, stream>>>(cachev, Vh);

  // 2) fused QKV projection (scatter into Q + KV-cache tail)
  k_gemm<0><<<dim3(3 * D_ / 64, MROWS_ / 128), 256, 0, stream>>>(
      Xh, Wqkvt, MROWS_, 3 * D_, D_, Qh, Kh, Vh, nullptr, nullptr);

  // 3) flash attention over prefix + causal chunk
  k_attn<<<dim3(B_ * H_, TQ_ / 64), 128, 0, stream>>>(Qh, Kh, Vh, Ctx);

  // 4) output projection + bias (fp32 out)
  k_gemm<1><<<dim3(D_ / 64, MROWS_ / 128), 256, 0, stream>>>(
      Ctx, Wprojt, MROWS_, D_, D_, nullptr, nullptr, nullptr, bproj, out);
}
